// LlamaAttention_39513699123202
// MI455X (gfx1250) — matrix operand order repaired and checked by reading, not compiled, not run on hardware
//
#include <hip/hip_runtime.h>
#include <math.h>

// Problem constants (match reference)
#define HDIM  4096
#define NHEAD 32
#define DHEAD 128
#define BATCH 2
#define SEQ   2048
#define MTOT  (BATCH * SEQ)   // 4096 rows

typedef __bf16 bf16;
typedef __bf16 v16bf __attribute__((ext_vector_type(16)));
typedef float  v8f   __attribute__((ext_vector_type(8)));

// ---------------------------------------------------------------------------
// gfx1250 async global->LDS DMA (ASYNCcnt-tracked), B128 per lane.
// GVS mode: SGPR64 base + per-lane u32 byte offset. LDS dest = low 32 bits of
// the flat address of a __shared__ location (flat LDS aperture convention).
// ---------------------------------------------------------------------------
__device__ __forceinline__ void async_b128(const void* gbase, unsigned gbyteoff,
                                           void* lds_ptr) {
  unsigned lo = (unsigned)(size_t)lds_ptr;
  asm volatile("global_load_async_to_lds_b128 %0, %1, %2"
               :: "v"(lo), "v"(gbyteoff), "s"(gbase)
               : "memory");
}
__device__ __forceinline__ void wait_async0() {
  asm volatile("s_wait_asynccnt 0" ::: "memory");
}

// ---------------------------------------------------------------------------
// fp32 -> bf16 cast (grid-stride)
// ---------------------------------------------------------------------------
__global__ void cast_f32_bf16(const float* __restrict__ src,
                              bf16* __restrict__ dst, int n) {
  int i = blockIdx.x * blockDim.x + threadIdx.x;
  int stride = gridDim.x * blockDim.x;
  for (; i < n; i += stride) dst[i] = (bf16)src[i];
}

// ---------------------------------------------------------------------------
// C[M,N] = A[M,K] x B[N,K]^T   (nn.Linear: x @ W.T), bf16 in, f32 accumulate.
// Block = 256 threads (8 waves). Tile: 128(M) x 128(N), K chunks of 32.
// Double-buffered async-DMA K-tiles; wave w owns rows [w*16, w*16+16).
// ---------------------------------------------------------------------------
__global__ __launch_bounds__(256) void gemm_wmma_bf16(
    const bf16* __restrict__ A, const bf16* __restrict__ B,
    float* __restrict__ Cf, bf16* __restrict__ Cb,
    int M, int N, int K)
{
  // row stride 40 elems = 80B: multiple of 16B (async B128 writes) and a
  // 20-bank step -> 16 distinct banks across 16 lanes on fragment reads.
  __shared__ bf16 As[2][128][40];
  __shared__ bf16 Bs[2][128][40];

  const int tid  = threadIdx.x;
  const int wave = tid >> 5;
  const int lane = tid & 31;
  const int half = lane >> 4;
  const int l16  = lane & 15;
  const int m0 = blockIdx.y * 128;
  const int n0 = blockIdx.x * 128;

  // Async tile fetch: 128x32 bf16 = 8KB per matrix = 2 B128 chunks per thread.
  auto load_tiles = [&](int kk, int buf) {
#pragma unroll
    for (int i = 0; i < 2; ++i) {
      int idx = tid + i * 256;          // 0..511
      int row = idx >> 2;               // 4 x 16B chunks per 64B row
      int ch  = idx & 3;
      unsigned aoff = (unsigned)((m0 + row) * K + kk) * 2u + (unsigned)ch * 16u;
      unsigned boff = (unsigned)((n0 + row) * K + kk) * 2u + (unsigned)ch * 16u;
      async_b128(A, aoff, &As[buf][row][ch * 8]);
      async_b128(B, boff, &Bs[buf][row][ch * 8]);
    }
  };

  v8f zero = {0.f,0.f,0.f,0.f,0.f,0.f,0.f,0.f};
  v8f acc[8];
#pragma unroll
  for (int i = 0; i < 8; ++i) acc[i] = zero;

  load_tiles(0, 0);
  wait_async0();
  __syncthreads();

  const int nk = K / 32;
  for (int t = 0; t < nk; ++t) {
    const int buf = t & 1;
    if (t + 1 < nk) load_tiles((t + 1) * 32, buf ^ 1);  // prefetch next tile

    // A fragment (contiguous k order: k = 16*half + e, matches B)
    v16bf af;
#pragma unroll
    for (int e = 0; e < 16; ++e) {
      int k = e + half * 16;
      af[e] = As[buf][wave * 16 + l16][k];
    }
    // Preload all 8 B fragments, then a back-to-back WMMA burst (XDL-friendly)
    v16bf bfr[8];
#pragma unroll
    for (int nt = 0; nt < 8; ++nt)
#pragma unroll
      for (int e = 0; e < 16; ++e)
        bfr[nt][e] = Bs[buf][nt * 16 + l16][e + half * 16];
#pragma unroll
    for (int nt = 0; nt < 8; ++nt)
      acc[nt] = __builtin_amdgcn_wmma_f32_16x16x32_bf16(
          false, af, false, bfr[nt], (short)0, acc[nt], false, false);

    wait_async0();     // next tile landed (overlapped with the 8 WMMAs above)
    __syncthreads();
  }

  // C/D layout: VGPR v -> row v + half*8, lane -> col
#pragma unroll
  for (int nt = 0; nt < 8; ++nt) {
#pragma unroll
    for (int v = 0; v < 8; ++v) {
      size_t r = (size_t)(m0 + wave * 16 + v + half * 8);
      size_t c = (size_t)(n0 + nt * 16 + l16);
      if (Cf) Cf[r * N + c] = acc[nt][v];
      else    Cb[r * N + c] = (bf16)acc[nt][v];
    }
  }
}

// ---------------------------------------------------------------------------
// RoPE on Q,K + repack [B,S,NH,D] -> [B,NH,S,D] (bf16). V repacked only.
// ---------------------------------------------------------------------------
__global__ void rope_pack(const bf16* __restrict__ Qt, const bf16* __restrict__ Kt,
                          const bf16* __restrict__ Vt, const int* __restrict__ past_lens,
                          bf16* __restrict__ Qb, bf16* __restrict__ Kb,
                          bf16* __restrict__ Vb)
{
  int idx = blockIdx.x * blockDim.x + threadIdx.x; // B*NH*S*64 threads
  int d = idx & 63;
  int s = (idx >> 6) & (SEQ - 1);
  int h = (idx >> 17) & (NHEAD - 1);
  int b = idx >> 22;
  if (b >= BATCH) return;

  float t   = (float)(past_lens[b] + s);
  float inv = __powf(10000.0f, -(float)(2 * d) / (float)DHEAD);
  float sn, cs;
  __sincosf(t * inv, &sn, &cs);

  size_t src = ((size_t)(b * SEQ + s)) * HDIM + (size_t)h * DHEAD;
  size_t dst = (((size_t)(b * NHEAD + h)) * SEQ + s) * DHEAD;

  float q0 = (float)Qt[src + d], q1 = (float)Qt[src + d + 64];
  float k0 = (float)Kt[src + d], k1 = (float)Kt[src + d + 64];
  Qb[dst + d]      = (bf16)(q0 * cs - q1 * sn);
  Qb[dst + d + 64] = (bf16)(q1 * cs + q0 * sn);
  Kb[dst + d]      = (bf16)(k0 * cs - k1 * sn);
  Kb[dst + d + 64] = (bf16)(k1 * cs + k0 * sn);
  Vb[dst + d]      = Vt[src + d];
  Vb[dst + d + 64] = Vt[src + d + 64];
}

// ---------------------------------------------------------------------------
// Causal flash attention: block = (128 q-rows, one (b,h)); 8 waves x 16 rows.
// Keys streamed in double-buffered async-DMA tiles of 32 with online softmax.
// ---------------------------------------------------------------------------
__global__ __launch_bounds__(256) void attn_wmma(
    const bf16* __restrict__ Qb, const bf16* __restrict__ Kb,
    const bf16* __restrict__ Vb, bf16* __restrict__ Ctx)
{
  // row stride 136 elems = 272B: multiple of 16B, 4-bank step -> 16 distinct
  // banks for the row-strided K fragment reads.
  __shared__ bf16 Ks[2][32][136];
  __shared__ bf16 Vs[2][32][136];
  __shared__ bf16 Ps[8][16][34];     // per-wave P staging (16x32)

  const int tid  = threadIdx.x;
  const int wave = tid >> 5;
  const int lane = tid & 31;
  const int half = lane >> 4;
  const int l16  = lane & 15;
  const int bh = blockIdx.y;                 // b*NH + h
  const int b  = bh / NHEAD;
  const int h  = bh % NHEAD;
  const int qt = blockIdx.x;                 // tile of 128 q rows
  const size_t base = (size_t)bh * SEQ * DHEAD;
  const unsigned base_b = (unsigned)base * 2u;   // byte offset (fits u32)
  const int qr0 = qt * 128 + wave * 16;

  // Async K/V tile fetch: 32x128 bf16 = 8KB each = 2 B128 chunks per thread.
  auto load_kv = [&](int j, int buf) {
#pragma unroll
    for (int i = 0; i < 2; ++i) {
      int idx = tid + i * 256;          // 0..511
      int row = idx >> 4;               // 16 x 16B chunks per 256B row
      int ch  = idx & 15;
      unsigned off = base_b + (unsigned)((j * 32 + row) * DHEAD) * 2u
                   + (unsigned)ch * 16u;
      async_b128(Kb, off, &Ks[buf][row][ch * 8]);
      async_b128(Vb, off, &Vs[buf][row][ch * 8]);
    }
  };

  // Q fragments: 16 rows x 128 (4 chunks of K=32), resident in VGPRs
  v16bf qf[4];
#pragma unroll
  for (int c4 = 0; c4 < 4; ++c4)
#pragma unroll
    for (int e = 0; e < 16; ++e) {
      int k = e + half * 16;
      qf[c4][e] = Qb[base + (size_t)(qr0 + l16) * DHEAD + c4 * 32 + k];
    }

  v8f zero = {0.f,0.f,0.f,0.f,0.f,0.f,0.f,0.f};
  v8f o[8];
#pragma unroll
  for (int i = 0; i < 8; ++i) o[i] = zero;
  float mrow[8], lrow[8];
#pragma unroll
  for (int v = 0; v < 8; ++v) { mrow[v] = -1e30f; lrow[v] = 0.f; }

  const float scale = 0.08838834764831845f;  // 1/sqrt(128)
  const int jmax = (qt + 1) * 4;             // key tiles of 32, causal bound

  load_kv(0, 0);
  wait_async0();
  __syncthreads();

  for (int j = 0; j < jmax; ++j) {
    const int buf = j & 1;
    if (j + 1 < jmax) load_kv(j + 1, buf ^ 1);   // prefetch next K/V tile

    // S = Q x K^T : 16x32 scores as two 16x16 WMMA accumulations over D
    v8f sc[2];
#pragma unroll
    for (int nt = 0; nt < 2; ++nt) {
      v8f a = zero;
#pragma unroll
      for (int c4 = 0; c4 < 4; ++c4) {
        v16bf bk;
#pragma unroll
        for (int e = 0; e < 16; ++e)
          bk[e] = Ks[buf][nt * 16 + l16][c4 * 32 + e + half * 16];
        a = __builtin_amdgcn_wmma_f32_16x16x32_bf16(
            false, qf[c4], false, bk, (short)0, a, false, false);
      }
      sc[nt] = a;
    }

    // scale + causal mask + online softmax (row spread over 16 lanes)
#pragma unroll
    for (int v = 0; v < 8; ++v) {
      int qrow = qr0 + v + half * 8;
      float m_t = -1e30f;
#pragma unroll
      for (int nt = 0; nt < 2; ++nt) {
        int key = j * 32 + nt * 16 + l16;
        float x = sc[nt][v] * scale;
        x = (key <= qrow) ? x : -1e30f;
        sc[nt][v] = x;
        m_t = fmaxf(m_t, x);
      }
#pragma unroll
      for (int sh = 8; sh >= 1; sh >>= 1)
        m_t = fmaxf(m_t, __shfl_xor(m_t, sh, 16));
      float m_new = fmaxf(mrow[v], m_t);
      float corr  = __expf(mrow[v] - m_new);
      mrow[v] = m_new;
      float rsum = 0.f;
#pragma unroll
      for (int nt = 0; nt < 2; ++nt) {
        float p = __expf(sc[nt][v] - m_new);
        rsum += p;
        Ps[wave][v + half * 8][nt * 16 + l16] = (bf16)p;
      }
#pragma unroll
      for (int sh = 8; sh >= 1; sh >>= 1)
        rsum += __shfl_xor(rsum, sh, 16);
      lrow[v] = lrow[v] * corr + rsum;
#pragma unroll
      for (int dt = 0; dt < 8; ++dt) o[dt][v] *= corr;
    }

    // O += P x V : P re-laid out as A fragment, V read transposed from LDS.
    // V fragments preloaded in groups of 4 for denser WMMA issue.
    v16bf pf;
#pragma unroll
    for (int e = 0; e < 16; ++e) {
      int k = e + half * 16;
      pf[e] = Ps[wave][l16][k];
    }
#pragma unroll
    for (int g = 0; g < 2; ++g) {
      v16bf bv[4];
#pragma unroll
      for (int i = 0; i < 4; ++i)
#pragma unroll
        for (int e = 0; e < 16; ++e)
          bv[i][e] = Vs[buf][e + half * 16][(g * 4 + i) * 16 + l16];
#pragma unroll
      for (int i = 0; i < 4; ++i)
        o[g * 4 + i] = __builtin_amdgcn_wmma_f32_16x16x32_bf16(
            false, pf, false, bv[i], (short)0, o[g * 4 + i], false, false);
    }

    wait_async0();     // next K/V tile landed (overlapped with compute above)
    __syncthreads();
  }

  // normalize and write context [b, s, h*D + d] (bf16 for O-projection GEMM)
#pragma unroll
  for (int v = 0; v < 8; ++v) {
    int s = qr0 + v + half * 8;
    float rl = 1.0f / lrow[v];
#pragma unroll
    for (int dt = 0; dt < 8; ++dt)
      Ctx[((size_t)(b * SEQ + s)) * HDIM + (size_t)h * DHEAD + dt * 16 + l16] =
          (bf16)(o[dt][v] * rl);
  }
}

// ---------------------------------------------------------------------------
extern "C" void kernel_launch(void* const* d_in, const int* in_sizes, int n_in,
                              void* d_out, int out_size, void* d_ws, size_t ws_size,
                              hipStream_t stream)
{
  (void)in_sizes; (void)n_in; (void)out_size; (void)ws_size;
  const float* hs = (const float*)d_in[0];
  const float* Wq = (const float*)d_in[1];
  const float* Wk = (const float*)d_in[2];
  const float* Wv = (const float*)d_in[3];
  const float* Wo = (const float*)d_in[4];
  const int* past = (const int*)d_in[5];

  const size_t NE = (size_t)MTOT * HDIM;   // 16,777,216 elements per matrix
  char* p = (char*)d_ws;
  bf16* Xb  = (bf16*)p; p += NE * sizeof(bf16);
  bf16* Wqb = (bf16*)p; p += NE * sizeof(bf16);
  bf16* Wkb = (bf16*)p; p += NE * sizeof(bf16);
  bf16* Wvb = (bf16*)p; p += NE * sizeof(bf16);
  bf16* Wob = (bf16*)p; p += NE * sizeof(bf16);
  bf16* Qt  = (bf16*)p; p += NE * sizeof(bf16);   // [B,S,H] pre-RoPE
  bf16* Kt  = (bf16*)p; p += NE * sizeof(bf16);
  bf16* Vt  = (bf16*)p; p += NE * sizeof(bf16);
  bf16* Qh  = (bf16*)p; p += NE * sizeof(bf16);   // [B,NH,S,D] post-RoPE
  bf16* Kh  = (bf16*)p; p += NE * sizeof(bf16);
  bf16* Vh  = (bf16*)p; p += NE * sizeof(bf16);
  bf16* Ctx = (bf16*)p; p += NE * sizeof(bf16);

  const int n = (int)NE;
  cast_f32_bf16<<<4096, 256, 0, stream>>>(hs, Xb, n);
  cast_f32_bf16<<<4096, 256, 0, stream>>>(Wq, Wqb, n);
  cast_f32_bf16<<<4096, 256, 0, stream>>>(Wk, Wkb, n);
  cast_f32_bf16<<<4096, 256, 0, stream>>>(Wv, Wvb, n);
  cast_f32_bf16<<<4096, 256, 0, stream>>>(Wo, Wob, n);

  dim3 gg(HDIM / 128, MTOT / 128), gb(256);
  gemm_wmma_bf16<<<gg, gb, 0, stream>>>(Xb, Wqb, nullptr, Qt, MTOT, HDIM, HDIM);
  gemm_wmma_bf16<<<gg, gb, 0, stream>>>(Xb, Wkb, nullptr, Kt, MTOT, HDIM, HDIM);
  gemm_wmma_bf16<<<gg, gb, 0, stream>>>(Xb, Wvb, nullptr, Vt, MTOT, HDIM, HDIM);

  int rope_threads = BATCH * NHEAD * SEQ * 64;
  rope_pack<<<rope_threads / 256, 256, 0, stream>>>(Qt, Kt, Vt, past, Qh, Kh, Vh);

  attn_wmma<<<dim3(SEQ / 128, BATCH * NHEAD), 256, 0, stream>>>(Qh, Kh, Vh, Ctx);

  gemm_wmma_bf16<<<gg, gb, 0, stream>>>(Ctx, Wob, (float*)d_out, nullptr,
                                        MTOT, HDIM, HDIM);
}
